// BEVFeatureAggregation_65326452572536
// MI455X (gfx1250) — matrix operand order repaired and checked by reading, not compiled, not run on hardware
//
#include <hip/hip_runtime.h>
#include <hip/hip_bf16.h>
#include <stdint.h>

// ---------------- problem constants (from reference setup_inputs) -----------
#define B_    4
#define N_    10000
#define C_    256
#define H_    200
#define W_    400
#define ROWS  (B_ * N_)        // 40000, multiple of 16
#define HW_   (H_ * W_)

typedef __attribute__((ext_vector_type(16))) __bf16 v16bf;
typedef __attribute__((ext_vector_type(8)))  float  v8f;

__device__ __forceinline__ unsigned short f32_to_bf16_rne(float f) {
    unsigned int u = __float_as_uint(f);
    unsigned int r = u + 0x7FFFu + ((u >> 16) & 1u);   // round-to-nearest-even
    return (unsigned short)(r >> 16);
}

// ---------------- kernel 1: W_proj f32 -> bf16 (row-major, (o,c)) ----------
// Row-major W_proj IS the WMMA B-matrix layout source: B[k][n] = W_proj[n][k],
// and each lane of a B fragment reads 16 contiguous K values of one row n.
__global__ void prep_w_kernel(const float* __restrict__ wp,
                              unsigned short* __restrict__ wbf) {
    int i = blockIdx.x * blockDim.x + threadIdx.x;
    if (i < C_ * C_) wbf[i] = f32_to_bf16_rne(wp[i]);
}

// ---------------- kernel 2: bilinear BEV gather -> feat bf16 (row-major) ---
__device__ __forceinline__ float corner_w(float xc, float yc, float w, int* off) {
    bool valid = (xc >= 0.0f) && (xc <= (float)(W_ - 1)) &&
                 (yc >= 0.0f) && (yc <= (float)(H_ - 1));
    int xi = (int)fminf(fmaxf(xc, 0.0f), (float)(W_ - 1));
    int yi = (int)fminf(fmaxf(yc, 0.0f), (float)(H_ - 1));
    *off = yi * W_ + xi;
    return valid ? w : 0.0f;
}

__global__ __launch_bounds__(256)
void sample_kernel(const float* __restrict__ anchor,
                   const float* __restrict__ bev,
                   unsigned short* __restrict__ feat) {
    int gtid = blockIdx.x * blockDim.x + threadIdx.x;
    int wave = gtid >> 5;                 // one wave per point
    int lane = threadIdx.x & 31;
    if (wave >= ROWS) return;
    int b = wave / N_;

    float ax = anchor[wave * 3 + 0];
    float ay = anchor[wave * 3 + 1];
    // grid_x (from ax) is the HEIGHT coord, grid_y (from ay) is the WIDTH coord
    float gxh = (ax + 80.0f) / (200.0f + 1e-6f) * 2.0f - 1.0f;
    float gyw = (ay + 40.0f) / (80.0f + 1e-6f) * 2.0f - 1.0f;
    float ix = (gyw + 1.0f) * 0.5f * (float)(W_ - 1);
    float iy = (gxh + 1.0f) * 0.5f * (float)(H_ - 1);

    float x0 = floorf(ix), y0 = floorf(iy);
    float x1 = x0 + 1.0f,  y1 = y0 + 1.0f;
    float wx1 = ix - x0, wx0 = 1.0f - wx1;
    float wy1 = iy - y0, wy0 = 1.0f - wy1;

    int o00, o10, o01, o11;
    float w00 = corner_w(x0, y0, wx0 * wy0, &o00);
    float w10 = corner_w(x1, y0, wx1 * wy0, &o10);
    float w01 = corner_w(x0, y1, wx0 * wy1, &o01);
    float w11 = corner_w(x1, y1, wx1 * wy1, &o11);

    const float* fb = bev + (size_t)b * C_ * HW_;
    union { unsigned short u[8]; uint4 q; } pk;
    int c0 = lane * 8;                     // 8 contiguous channels per lane
#pragma unroll
    for (int i = 0; i < 8; ++i) {
        const float* fc = fb + (size_t)(c0 + i) * HW_;
        float v = w00 * fc[o00] + w10 * fc[o10] + w01 * fc[o01] + w11 * fc[o11];
        pk.u[i] = f32_to_bf16_rne(v);
    }
    // coalesced 16B store; feat row-major (ROWS x C_) bf16
    *(uint4*)(feat + (size_t)wave * C_ + c0) = pk.q;
}

// ---------------- kernel 3: WMMA GEMM + bias + residual ---------------------
// Each wave: 16 rows x 64 cols of output (4 accumulators), K=256 in 8 steps.
union FragBF { v16bf v; uint4 q[2]; };

__global__ __launch_bounds__(256)
void gemm_kernel(const unsigned short* __restrict__ feat,
                 const unsigned short* __restrict__ wbf,
                 const float* __restrict__ bias,
                 const float* __restrict__ inst,
                 float* __restrict__ out) {
    int gtid = blockIdx.x * blockDim.x + threadIdx.x;
    int wave = gtid >> 5;
    int lane = threadIdx.x & 31;
    const int tasks = (ROWS / 16) * (C_ / 64);      // 10000 wave-tiles
    if (wave >= tasks) return;

    int rowTile = wave >> 2;            // / (C_/64)
    int colGrp  = wave & 3;
    int m0 = rowTile * 16;
    int n0 = colGrp * 64;
    int l15 = lane & 15;
    int hi  = (lane >> 4) & 1;          // which half-wave

    v8f acc[4] = {v8f{}, v8f{}, v8f{}, v8f{}};

    // A: row (m0 + l15); per k-step lanes<16 read K[0..15],
    // lanes>=16 read K[16..31]  (same k order as B)
    const unsigned short* arow = feat + (size_t)(m0 + l15) * C_;
    // B: column n of B == row n of W_proj; lanes<16 read K[0..15], lanes>=16 K[16..31]
    const unsigned short* brow0 = wbf + (size_t)(n0 +  0 + l15) * C_;
    const unsigned short* brow1 = wbf + (size_t)(n0 + 16 + l15) * C_;
    const unsigned short* brow2 = wbf + (size_t)(n0 + 32 + l15) * C_;
    const unsigned short* brow3 = wbf + (size_t)(n0 + 48 + l15) * C_;

#pragma unroll
    for (int kt = 0; kt < 8; ++kt) {
        int ka = kt * 32 + hi * 16;
        FragA_load:;
        FragBF a;
        a.q[0] = *(const uint4*)(arow + ka);
        a.q[1] = *(const uint4*)(arow + ka + 8);

        int kb = kt * 32 + hi * 16;
        FragBF b0, b1, b2, b3;
        b0.q[0] = *(const uint4*)(brow0 + kb); b0.q[1] = *(const uint4*)(brow0 + kb + 8);
        b1.q[0] = *(const uint4*)(brow1 + kb); b1.q[1] = *(const uint4*)(brow1 + kb + 8);
        b2.q[0] = *(const uint4*)(brow2 + kb); b2.q[1] = *(const uint4*)(brow2 + kb + 8);
        b3.q[0] = *(const uint4*)(brow3 + kb); b3.q[1] = *(const uint4*)(brow3 + kb + 8);

        acc[0] = __builtin_amdgcn_wmma_f32_16x16x32_bf16(false, a.v, false, b0.v, (short)0, acc[0], false, false);
        acc[1] = __builtin_amdgcn_wmma_f32_16x16x32_bf16(false, a.v, false, b1.v, (short)0, acc[1], false, false);
        acc[2] = __builtin_amdgcn_wmma_f32_16x16x32_bf16(false, a.v, false, b2.v, (short)0, acc[2], false, false);
        acc[3] = __builtin_amdgcn_wmma_f32_16x16x32_bf16(false, a.v, false, b3.v, (short)0, acc[3], false, false);
    }

    // Epilogue: D(m,n) lives in VGPR (m&7), lane (m>=8?16:0)+n.
#pragma unroll
    for (int t = 0; t < 4; ++t) {
        int col = n0 + t * 16 + l15;
        float bv = bias[col];
#pragma unroll
        for (int v = 0; v < 8; ++v) {
            int m = m0 + v + hi * 8;
            size_t idx = (size_t)m * C_ + col;
            out[idx] = acc[t][v] + bv + inst[idx];
        }
    }
}

// ---------------- host-side launcher ---------------------------------------
extern "C" void kernel_launch(void* const* d_in, const int* in_sizes, int n_in,
                              void* d_out, int out_size, void* d_ws, size_t ws_size,
                              hipStream_t stream) {
    const float* inst   = (const float*)d_in[0];   // (B,N,C)
    const float* anchor = (const float*)d_in[1];   // (B,N,3)
    // d_in[2] = anchor_embed: unused by the reference computation
    const float* bev    = (const float*)d_in[3];   // (B,C,H,W)
    const float* wp     = (const float*)d_in[4];   // (C,C)
    const float* bias   = (const float*)d_in[5];   // (C,)
    float* out = (float*)d_out;

    // workspace: feat bf16 (ROWS*C) then W bf16 (C*C); ~20.6 MB total
    unsigned short* feat = (unsigned short*)d_ws;
    unsigned short* wbf  = feat + (size_t)ROWS * C_;

    hipLaunchKernelGGL(prep_w_kernel, dim3((C_ * C_ + 255) / 256), dim3(256), 0, stream,
                       wp, wbf);

    int sampleBlocks = (ROWS * 32 + 255) / 256;    // one wave per point
    hipLaunchKernelGGL(sample_kernel, dim3(sampleBlocks), dim3(256), 0, stream,
                       anchor, bev, feat);

    int tasks = (ROWS / 16) * (C_ / 64);           // 10000 wave-tiles
    int gemmBlocks = (tasks * 32 + 255) / 256;
    hipLaunchKernelGGL(gemm_kernel, dim3(gemmBlocks), dim3(256), 0, stream,
                       feat, wbf, bias, inst, out);
}
